// PPO_Actor_33569464385629
// MI455X (gfx1250) — matrix operand order repaired and checked by reading, not compiled, not run on hardware
//
#include <hip/hip_runtime.h>

// ---------------------------------------------------------------------------
// Fused PPO actor for MI455X (gfx1250, wave32, WMMA 16x16x32 bf16).
//   x[B,N,O] --MLP--> feats --per-agent LSTM--> h,c[N,B,H] --head--> logits[B,N,A]
// B=4096 N=32 O=128 F=64 H=64 A=16
// One block = one agent x 128 batch rows; 8 waves, each wave = 16 rows.
// Weights staged f32->bf16 into LDS in *fragment-packed* B layout so each
// WMMA B operand is two contiguous ds_load_b128 per lane.
// ---------------------------------------------------------------------------

typedef __attribute__((ext_vector_type(16))) __bf16 v16bf;
typedef __attribute__((ext_vector_type(8)))  __bf16 v8bf;
typedef __attribute__((ext_vector_type(4)))  __bf16 v4bf;
typedef __attribute__((ext_vector_type(8)))  float  v8f;

constexpr int Bc = 4096, Nc = 32, Oc = 128, Fc = 64, Hc = 64, Ac = 16;
constexpr int TILE_B = 128;   // batch rows per block
constexpr int WAVES  = 8;     // 8 waves * 16 rows = 128 rows

// ---- A fragment: 16x32 bf16, row-major source buf[m*ld + k] ---------------
// ISA 7.12.2: lanes 0-15 m=0..15 hold K {0..7,16..23}; lanes 16-31 hold
// K {8..15,24..31}.  Two contiguous 16-byte LDS reads per lane.
__device__ __forceinline__ v16bf load_a_frag(const __bf16* buf, int ld,
                                             int kbase, int lane) {
  const int m    = lane & 15;
  const int koff = ((lane >> 4) << 3) + kbase;
  const __bf16* p = buf + m * ld + koff;
  v8bf lo = *(const v8bf*)(p);        // K = koff .. koff+7
  v8bf hi = *(const v8bf*)(p + 16);   // K = koff+16 .. koff+23
  return __builtin_shufflevector(lo, hi, 0, 1, 2, 3, 4, 5, 6, 7,
                                 8, 9, 10, 11, 12, 13, 14, 15);
}

// ---- B fragment from fragment-packed LDS ----------------------------------
// packed[((tile*KSTEPS + ks)*32 + lane)*16 + e] = W[n][k] with
//   n = tile*16 + (lane&15),  k = ks*32 + ((lane>>4)<<3) + (e&7) + ((e>>3)<<4)
// -> per lane: 32 contiguous bytes, two ds_load_b128.
template <int KSTEPS>
__device__ __forceinline__ v16bf load_b_packed(const __bf16* base, int tile,
                                               int ks, int lane) {
  const __bf16* p = base + (((tile * KSTEPS + ks) * 32 + lane) << 4);
  v8bf lo = *(const v8bf*)(p);
  v8bf hi = *(const v8bf*)(p + 8);
  return __builtin_shufflevector(lo, hi, 0, 1, 2, 3, 4, 5, 6, 7,
                                 8, 9, 10, 11, 12, 13, 14, 15);
}

// ---- stage one weight matrix W[NCOLS][K] (row-major f32) into packed bf16 -
template <int K, int NCOLS>
__device__ __forceinline__ void stage_weight_packed(__bf16* dst,
                                                    const float* __restrict__ src,
                                                    int tid) {
  constexpr int KSTEPS = K / 32;
  constexpr int T8 = (NCOLS * K) / 8;
  for (int i8 = tid; i8 < T8; i8 += 256) {
    const int i      = i8 << 3;
    const int e0     = i & 15;
    const int lane_l = (i >> 4) & 31;
    const int f      = i >> 9;
    const int ks     = f % KSTEPS;
    const int tile   = f / KSTEPS;
    const int nn     = tile * 16 + (lane_l & 15);
    const int k0     = ((lane_l >> 4) << 3) + ks * 32 + ((e0 >> 3) << 4);
    const float* s   = src + nn * K + k0;       // 8 contiguous floats
    v8bf v;
#pragma unroll
    for (int e = 0; e < 8; ++e) v[e] = (__bf16)s[e];
    *(v8bf*)(dst + i) = v;
  }
}

__device__ __forceinline__ float sigmoidf_(float v) {
  return 1.0f / (1.0f + __expf(-v));
}

__global__ __launch_bounds__(256, 1)
void ppo_actor_fused(const float* __restrict__ x,
                     const float* __restrict__ h0,
                     const float* __restrict__ c0,
                     const float* __restrict__ W1,   const float* __restrict__ b1,
                     const float* __restrict__ W2,   const float* __restrict__ b2,
                     const float* __restrict__ W_ih, const float* __restrict__ b_ih,
                     const float* __restrict__ W_hh, const float* __restrict__ b_hh,
                     const float* __restrict__ W_hd, const float* __restrict__ b_hd,
                     float* __restrict__ out_logits,
                     float* __restrict__ out_h,
                     float* __restrict__ out_c) {
  // Fragment-packed bf16 weights in LDS.
  __shared__ __align__(16) __bf16 pW1 [Oc * 64];        // 16 KB
  __shared__ __align__(16) __bf16 pW2 [64 * Fc];        //  8 KB
  __shared__ __align__(16) __bf16 pWih[Fc * 4 * Hc];    // 32 KB
  __shared__ __align__(16) __bf16 pWhh[Hc * 4 * Hc];    // 32 KB
  __shared__ __align__(16) __bf16 pWhd[Hc * Ac];        //  2 KB
  __shared__ __align__(16) __bf16 sAct[WAVES][16 * Oc]; // 32 KB (per-wave 4 KB)

  const int tid  = threadIdx.x;
  const int lane = tid & 31;
  const int wave = tid >> 5;
  const int n    = blockIdx.y;                       // agent index
  const int rowbase = blockIdx.x * TILE_B + wave * 16;

  // ---------------- stage weights (whole block, vectorized) ----------------
  stage_weight_packed<Oc, 64>(pW1, W1, tid);                       // W1 [64][128]
  stage_weight_packed<64, Fc>(pW2, W2, tid);                       // W2 [64][64]
  stage_weight_packed<Fc, 4 * Hc>(pWih, W_ih + (size_t)n * 4 * Hc * Fc, tid);
  stage_weight_packed<Hc, 4 * Hc>(pWhh, W_hh + (size_t)n * 4 * Hc * Hc, tid);
  stage_weight_packed<Hc, Ac>(pWhd, W_hd, tid);                    // W_head [16][64]
  __syncthreads();

  __bf16* buf  = &sAct[wave][0];     // 16 x 128 staging for x
  __bf16* bufA = buf;                // 16 x 64 region A (ld 64)
  __bf16* bufB = buf + 16 * 64;      // 16 x 64 region B (ld 64)

  // ---------------- stage x tile: 16 rows x 128 obs (float4 reads) ---------
  {
    const float* xs = x + (size_t)n * Oc;            // + b*N*O per row
#pragma unroll 4
    for (int i4 = lane; i4 < (16 * Oc) / 4; i4 += 32) {
      const int i = i4 << 2;
      const int r = i >> 7, k = i & (Oc - 1);
      const float* s = xs + (size_t)(rowbase + r) * (Nc * Oc) + k;
      v4bf v;
#pragma unroll
      for (int e = 0; e < 4; ++e) v[e] = (__bf16)s[e];
      *(v4bf*)(buf + i) = v;
    }
  }

  const int col     = lane & 15;
  const int halfsel = lane >> 4;

  // ---------------- MLP1: [16x128] @ [128x64], relu+b1 ---------------------
  v8f acc1[4] = {};
#pragma unroll
  for (int ks = 0; ks < 4; ++ks) {
    v16bf a = load_a_frag(buf, Oc, ks * 32, lane);
#pragma unroll
    for (int t = 0; t < 4; ++t) {
      v16bf b = load_b_packed<4>(pW1, t, ks, lane);
      acc1[t] = __builtin_amdgcn_wmma_f32_16x16x32_bf16(
          false, a, false, b, (short)0, acc1[t], false, false);
    }
  }
#pragma unroll
  for (int t = 0; t < 4; ++t) {
    float bb = b1[t * 16 + col];
#pragma unroll
    for (int r = 0; r < 8; ++r) {
      float v = acc1[t][r] + bb;
      v = v > 0.0f ? v : 0.0f;
      bufA[(r + halfsel * 8) * 64 + t * 16 + col] = (__bf16)v;  // f1 -> A
    }
  }

  // ---------------- MLP2: [16x64] @ [64x64], relu+b2 -----------------------
  v8f acc2[4] = {};
#pragma unroll
  for (int ks = 0; ks < 2; ++ks) {
    v16bf a = load_a_frag(bufA, 64, ks * 32, lane);
#pragma unroll
    for (int t = 0; t < 4; ++t) {
      v16bf b = load_b_packed<2>(pW2, t, ks, lane);
      acc2[t] = __builtin_amdgcn_wmma_f32_16x16x32_bf16(
          false, a, false, b, (short)0, acc2[t], false, false);
    }
  }
#pragma unroll
  for (int t = 0; t < 4; ++t) {
    float bb = b2[t * 16 + col];
#pragma unroll
    for (int r = 0; r < 8; ++r) {
      float v = acc2[t][r] + bb;
      v = v > 0.0f ? v : 0.0f;
      bufB[(r + halfsel * 8) * 64 + t * 16 + col] = (__bf16)v;  // feats -> B
    }
  }

  // ---------------- LSTM gates: feats@W_ih^T + h0@W_hh^T -------------------
  v8f g[16] = {};                       // 16 tiles of 16 cols = 256 gate cols
#pragma unroll
  for (int ks = 0; ks < 2; ++ks) {
    v16bf a = load_a_frag(bufB, 64, ks * 32, lane);  // feats
#pragma unroll
    for (int t = 0; t < 16; ++t) {
      v16bf b = load_b_packed<2>(pWih, t, ks, lane);
      g[t] = __builtin_amdgcn_wmma_f32_16x16x32_bf16(
          false, a, false, b, (short)0, g[t], false, false);
    }
  }
  {                                                   // stage h0 tile -> A
    const float* hs = h0 + (size_t)n * Bc * Hc;
#pragma unroll 4
    for (int i4 = lane; i4 < (16 * Hc) / 4; i4 += 32) {
      const int i = i4 << 2;
      const int r = i >> 6, k = i & 63;
      const float* s = hs + (size_t)(rowbase + r) * Hc + k;
      v4bf v;
#pragma unroll
      for (int e = 0; e < 4; ++e) v[e] = (__bf16)s[e];
      *(v4bf*)(bufA + i) = v;
    }
  }
#pragma unroll
  for (int ks = 0; ks < 2; ++ks) {
    v16bf a = load_a_frag(bufA, 64, ks * 32, lane);  // h0
#pragma unroll
    for (int t = 0; t < 16; ++t) {
      v16bf b = load_b_packed<2>(pWhh, t, ks, lane);
      g[t] = __builtin_amdgcn_wmma_f32_16x16x32_bf16(
          false, a, false, b, (short)0, g[t], false, false);
    }
  }

  // ---------------- elementwise LSTM cell + write h_new/c_new --------------
  const float* bihn = b_ih + (size_t)n * 4 * Hc;
  const float* bhhn = b_hh + (size_t)n * 4 * Hc;
  const float* c0n  = c0   + (size_t)n * Bc * Hc;
  float* hOut = out_h + (size_t)n * Bc * Hc;
  float* cOut = out_c + (size_t)n * Bc * Hc;

#pragma unroll
  for (int j = 0; j < 4; ++j) {                     // 4 tiles across H=64
    const int hcol = j * 16 + col;
    const float b_i = bihn[0 * Hc + hcol] + bhhn[0 * Hc + hcol];
    const float b_f = bihn[1 * Hc + hcol] + bhhn[1 * Hc + hcol];
    const float b_g = bihn[2 * Hc + hcol] + bhhn[2 * Hc + hcol];
    const float b_o = bihn[3 * Hc + hcol] + bhhn[3 * Hc + hcol];
#pragma unroll
    for (int r = 0; r < 8; ++r) {
      const int brow = rowbase + r + halfsel * 8;
      const float ig = sigmoidf_(g[0 * 4 + j][r] + b_i);
      const float fg = sigmoidf_(g[1 * 4 + j][r] + b_f);
      const float gg = tanhf(g[2 * 4 + j][r] + b_g);
      const float og = sigmoidf_(g[3 * 4 + j][r] + b_o);
      const float cprev = c0n[(size_t)brow * Hc + hcol];
      const float cn = fg * cprev + ig * gg;
      const float hn = og * tanhf(cn);
      cOut[(size_t)brow * Hc + hcol] = cn;
      hOut[(size_t)brow * Hc + hcol] = hn;
      bufA[(r + halfsel * 8) * 64 + hcol] = (__bf16)hn;   // h_new -> A
    }
  }

  // ---------------- action head: [16x64] @ [64x16] + b_head ----------------
  v8f la = {};
#pragma unroll
  for (int ks = 0; ks < 2; ++ks) {
    v16bf a = load_a_frag(bufA, 64, ks * 32, lane);
    v16bf b = load_b_packed<2>(pWhd, 0, ks, lane);
    la = __builtin_amdgcn_wmma_f32_16x16x32_bf16(
        false, a, false, b, (short)0, la, false, false);
  }
  const float bh = b_hd[col];
#pragma unroll
  for (int r = 0; r < 8; ++r) {
    const int brow = rowbase + r + halfsel * 8;
    out_logits[((size_t)brow * Nc + n) * Ac + col] = la[r] + bh;
  }
}

extern "C" void kernel_launch(void* const* d_in, const int* in_sizes, int n_in,
                              void* d_out, int out_size, void* d_ws, size_t ws_size,
                              hipStream_t stream) {
  (void)in_sizes; (void)n_in; (void)out_size; (void)d_ws; (void)ws_size;
  const float* x      = (const float*)d_in[0];
  const float* h0     = (const float*)d_in[1];
  const float* c0     = (const float*)d_in[2];
  const float* W1     = (const float*)d_in[3];
  const float* b1     = (const float*)d_in[4];
  const float* W2     = (const float*)d_in[5];
  const float* b2     = (const float*)d_in[6];
  const float* W_ih   = (const float*)d_in[7];
  const float* b_ih   = (const float*)d_in[8];
  const float* W_hh   = (const float*)d_in[9];
  const float* b_hh   = (const float*)d_in[10];
  const float* W_head = (const float*)d_in[11];
  const float* b_head = (const float*)d_in[12];

  float* out        = (float*)d_out;
  float* out_logits = out;                               // [B,N,A]
  float* out_h      = out + (size_t)Bc * Nc * Ac;        // [N,B,H]
  float* out_c      = out_h + (size_t)Nc * Bc * Hc;      // [N,B,H]

  dim3 grid(Bc / TILE_B, Nc);
  ppo_actor_fused<<<grid, 256, 0, stream>>>(
      x, h0, c0, W1, b1, W2, b2, W_ih, b_ih, W_hh, b_hh, W_head, b_head,
      out_logits, out_h, out_c);
}
